// Attention_79431125172941
// MI455X (gfx1250) — matrix operand order repaired and checked by reading, not compiled, not run on hardware
//
#include <hip/hip_runtime.h>
#include <hip/hip_bf16.h>

// ---------------------------------------------------------------------------
// MHSA for MI455X (gfx1250): wave32, v_wmma_f32_16x16x32_f16, async global->LDS
// DMA (ASYNCcnt) with double-buffered LDS in the GEMMs, flash attention with
// per-wave online softmax.
//   P0: one-shot f16 conversion of x, and f16 transpose of w_qkv / w_out
//   K1: QKV projection  -> q (pre-scaled), k, v^T  (f16, head-major)
//   K2: flash attention -> inner f16 [b*n][h*64+d]
//   K3: output projection + bias -> fp32 d_out
// ---------------------------------------------------------------------------

typedef _Float16 f16;
typedef __attribute__((ext_vector_type(4)))  _Float16 v4h;
typedef __attribute__((ext_vector_type(8)))  _Float16 v8h;
typedef __attribute__((ext_vector_type(16))) _Float16 v16h;
typedef __attribute__((ext_vector_type(4)))  float    v4f;
typedef __attribute__((ext_vector_type(8)))  float    v8f;

union H16 { v16h v; v8h h[2]; };

#define BM 128
#define BN 128
#define BK 32
#define LDS_STR 40            // f16 elems/row: 80B rows, keeps b128 alignment
#define TILE_B (128 * LDS_STR * 2)  // bytes per staged tile

__device__ __forceinline__ v8f wmma16(v16h a, v16h b, v8f c) {
  return __builtin_amdgcn_wmma_f32_16x16x32_f16(false, a, false, b, (short)0, c,
                                                false, false);
}

// A-fragment (16x32, row-major, ld in elems): lanes 0-15 row=lane K=[0..7]+[16..23],
// lanes 16-31 row=lane-16 K=[8..15]+[24..31]  -> two b128 loads per lane.
__device__ __forceinline__ v16h load_a_frag(const f16* base, int ld) {
  const int lane = threadIdx.x & 31;
  const int hal  = lane >> 4;
  const int row  = lane & 15;
  const f16* p = base + row * ld + hal * 8;
  H16 r;
  r.h[0] = *(const v8h*)(p);
  r.h[1] = *(const v8h*)(p + 16);
  return r.v;
}

// B-fragment (32x16) from "N rows contiguous along K" storage:
// lanes 0-15: col=lane K=[0..7]+[16..23]; lanes 16-31: col=lane-16 K=[8..15]+[24..31].
__device__ __forceinline__ v16h load_b_frag(const f16* base, int ld) {
  const int lane = threadIdx.x & 31;
  const int hal  = lane >> 4;
  const int n    = lane & 15;
  const f16* p = base + n * ld + hal * 8;
  H16 r;
  r.h[0] = *(const v8h*)(p);
  r.h[1] = *(const v8h*)(p + 16);
  return r.v;
}

// CDNA5 async DMA: 16B global -> LDS, tracked on ASYNCcnt (no VGPR landing).
__device__ __forceinline__ void async_copy16(unsigned lds_off, const f16* gptr) {
  unsigned long long ga = (unsigned long long)gptr;
  asm volatile("global_load_async_to_lds_b128 %0, %1, off"
               :: "v"(lds_off), "v"(ga) : "memory");
}

// ---------------------------------------------------------------------------
// Shared async double-buffered GEMM main loop: C += A[8192xK] * BT[n][K]^T
// (both operands f16, K=1024). 8 waves, each 32x64 sub-tile, 8 WMMA / K-step.
// ---------------------------------------------------------------------------
__device__ __forceinline__ void gemm_loop(const f16* __restrict__ A,
                                          const f16* __restrict__ BT,
                                          f16* smem, int mBase, int nBase,
                                          v8f acc[2][4]) {
  const int tid  = threadIdx.x;
  const int wave = tid >> 5;
  const int wm = wave & 3, wn = wave >> 2;
  const unsigned smem_off = (unsigned)(unsigned long long)smem;

  auto issue_tile = [&](int k0, int buf) {
#pragma unroll
    for (int j = 0; j < 2; ++j) {
      const int chunk = tid * 2 + j;       // 512 chunks of 16B per tile
      const int row = chunk >> 2;          // 4 chunks per 64B row
      const int seg = (chunk & 3) * 8;     // f16 elems
      const unsigned lo = smem_off + (unsigned)buf * 2u * TILE_B +
                          (unsigned)row * (LDS_STR * 2) + (unsigned)seg * 2;
      async_copy16(lo, A + (size_t)(mBase + row) * 1024 + k0 + seg);
      async_copy16(lo + TILE_B, BT + (size_t)(nBase + row) * 1024 + k0 + seg);
    }
  };

  issue_tile(0, 0);
  for (int it = 0; it < 1024 / BK; ++it) {
    if (it + 1 < 1024 / BK) {
      issue_tile((it + 1) * BK, (it + 1) & 1);
      asm volatile("s_wait_asynccnt 4" ::: "memory");  // tile `it` has landed
    } else {
      asm volatile("s_wait_asynccnt 0" ::: "memory");
    }
    __syncthreads();

    const f16* As = smem + (size_t)(it & 1) * (2 * 128 * LDS_STR);
    const f16* Bs = As + 128 * LDS_STR;
    const v16h af0 = load_a_frag(&As[(wm * 32 + 0)  * LDS_STR], LDS_STR);
    const v16h af1 = load_a_frag(&As[(wm * 32 + 16) * LDS_STR], LDS_STR);
#pragma unroll
    for (int sn = 0; sn < 4; ++sn) {
      const v16h bf = load_b_frag(&Bs[(wn * 64 + sn * 16) * LDS_STR], LDS_STR);
      acc[0][sn] = wmma16(af0, bf, acc[0][sn]);
      acc[1][sn] = wmma16(af1, bf, acc[1][sn]);
    }
    __syncthreads();  // reads done before this buffer is DMA'd again
  }
}

// ---------------------------------------------------------------------------
// P0a: bulk fp32 -> f16 (x, 8,388,608 elems; 4/thread)
// ---------------------------------------------------------------------------
__global__ __launch_bounds__(256)
void cvt_kernel(const float* __restrict__ in, f16* __restrict__ out) {
  const size_t i = ((size_t)blockIdx.x * 256 + threadIdx.x) * 4;
  v4f f = *(const v4f*)(in + i);
  v4h h = {(f16)f.x, (f16)f.y, (f16)f.z, (f16)f.w};
  *(v4h*)(out + i) = h;
}

// P0b: w[K][N] fp32 -> out[N][K] f16 via 32x32 LDS tile transpose
__global__ __launch_bounds__(256)
void transpose_cvt_kernel(const float* __restrict__ in, f16* __restrict__ out,
                          int K, int N) {
  __shared__ f16 tile[32][33];
  const int kt = blockIdx.y * 32, nt = blockIdx.x * 32;
  const int r  = threadIdx.x >> 3;
  const int c4 = (threadIdx.x & 7) * 4;
  v4f f = *(const v4f*)(in + (size_t)(kt + r) * N + nt + c4);
  tile[r][c4 + 0] = (f16)f.x;
  tile[r][c4 + 1] = (f16)f.y;
  tile[r][c4 + 2] = (f16)f.z;
  tile[r][c4 + 3] = (f16)f.w;
  __syncthreads();
  v4h o = {tile[c4 + 0][r], tile[c4 + 1][r], tile[c4 + 2][r], tile[c4 + 3][r]};
  *(v4h*)(out + (size_t)(nt + r) * K + kt + c4) = o;
}

// ---------------------------------------------------------------------------
// K1: QKV GEMM, epilogue scatters f16 into q (x SCALE), k, v^T head-major.
// ---------------------------------------------------------------------------
__global__ __launch_bounds__(256)
void qkv_gemm_kernel(const f16* __restrict__ A, const f16* __restrict__ BT,
                     f16* __restrict__ qb, f16* __restrict__ kb,
                     f16* __restrict__ vtb) {
  __shared__ f16 smem[4 * 128 * LDS_STR];  // A0,B0,A1,B1
  const int lane = threadIdx.x & 31;
  const int wave = threadIdx.x >> 5;
  const int wm = wave & 3, wn = wave >> 2;
  const int mBase = blockIdx.y * BM, nBase = blockIdx.x * BN;

  v8f acc[2][4] = {};
  gemm_loop(A, BT, smem, mBase, nBase, acc);

  const float scale = 0.03125f;  // DIM^-0.5 = 1/32, folded into Q
  const int hal = lane >> 4, ln = lane & 15;
#pragma unroll
  for (int sm = 0; sm < 2; ++sm)
#pragma unroll
    for (int sn = 0; sn < 4; ++sn)
#pragma unroll
      for (int r = 0; r < 8; ++r) {
        const int grow = mBase + wm * 32 + sm * 16 + r + hal * 8;
        const int gcol = nBase + wn * 64 + sn * 16 + ln;
        const float v = acc[sm][sn][r];
        const int which = gcol >> 10;  // 0=q 1=k 2=v (uniform per block)
        const int inr   = gcol & 1023;
        const int h = inr >> 6, d = inr & 63;
        const int b = grow >> 11, n = grow & 2047;
        const int bh = b * 16 + h;
        if (which == 0)      qb[((size_t)bh * 2048 + n) * 64 + d]  = (f16)(v * scale);
        else if (which == 1) kb[((size_t)bh * 2048 + n) * 64 + d]  = (f16)v;
        else                 vtb[((size_t)bh * 64 + d) * 2048 + n] = (f16)v;
      }
}

// ---------------------------------------------------------------------------
// K2: flash attention. Wave owns 16 query rows of one (b,h); 2048 keys in
// chunks of 32, online softmax; P flips C->A layout via per-wave LDS tile.
// ---------------------------------------------------------------------------
__global__ __launch_bounds__(256)
void attn_kernel(const f16* __restrict__ q, const f16* __restrict__ k,
                 const f16* __restrict__ vt, f16* __restrict__ inner) {
  __shared__ f16 Pbuf[8 * 16 * LDS_STR];

  const int tid = threadIdx.x, wave = tid >> 5, lane = tid & 31;
  const int hal = lane >> 4, ln = lane & 15;
  const int bh = blockIdx.x;
  const int b = bh >> 4, h = bh & 15;
  const int rBase = blockIdx.y * 128 + wave * 16;

  const f16* qrow = q + ((size_t)bh * 2048 + rBase) * 64;
  const v16h qf0 = load_a_frag(qrow, 64);
  const v16h qf1 = load_a_frag(qrow + 32, 64);

  v8f o[4] = {};
  float m[8], l[8];
#pragma unroll
  for (int r = 0; r < 8; ++r) { m[r] = -1e30f; l[r] = 0.0f; }

  f16* Pw = &Pbuf[wave * 16 * LDS_STR];

  for (int c0 = 0; c0 < 2048; c0 += 32) {
    if (c0 + 32 < 2048) {  // global_prefetch_b8 next K / V chunk
      __builtin_prefetch((const void*)(k + ((size_t)bh * 2048 + c0 + 32 + (lane >> 1)) * 64), 0, 1);
      __builtin_prefetch((const void*)(vt + ((size_t)bh * 64 + lane) * 2048 + c0 + 32), 0, 1);
    }
    // ---- S(16x32) = Q @ K^T
    v8f s[2];
#pragma unroll
    for (int nt = 0; nt < 2; ++nt) {
      const f16* kp = k + ((size_t)bh * 2048 + c0 + nt * 16) * 64;
      v8f z = {};
      z     = wmma16(qf0, load_b_frag(kp, 64), z);
      s[nt] = wmma16(qf1, load_b_frag(kp + 32, 64), z);
    }
    // ---- online softmax stats (rows live in vgpr index + lane-half)
    float alpha[8];
#pragma unroll
    for (int r = 0; r < 8; ++r) {
      float pm = fmaxf(s[0][r], s[1][r]);
      pm = fmaxf(pm, __shfl_xor(pm, 1, 16));
      pm = fmaxf(pm, __shfl_xor(pm, 2, 16));
      pm = fmaxf(pm, __shfl_xor(pm, 4, 16));
      pm = fmaxf(pm, __shfl_xor(pm, 8, 16));
      const float mnew = fmaxf(m[r], pm);
      alpha[r] = __expf(m[r] - mnew);
      m[r] = mnew;
    }
#pragma unroll
    for (int dt = 0; dt < 4; ++dt)
#pragma unroll
      for (int r = 0; r < 8; ++r) o[dt][r] = o[dt][r] * alpha[r];

    // ---- P = exp(S - m), row sums, spill P(f16) to LDS
#pragma unroll
    for (int r = 0; r < 8; ++r) {
      const float p0 = __expf(s[0][r] - m[r]);
      const float p1 = __expf(s[1][r] - m[r]);
      float ps = p0 + p1;
      ps += __shfl_xor(ps, 1, 16);
      ps += __shfl_xor(ps, 2, 16);
      ps += __shfl_xor(ps, 4, 16);
      ps += __shfl_xor(ps, 8, 16);
      l[r] = l[r] * alpha[r] + ps;
      const int prow = r + hal * 8;
      Pw[prow * LDS_STR + ln]      = (f16)p0;
      Pw[prow * LDS_STR + 16 + ln] = (f16)p1;
    }
    asm volatile("s_wait_dscnt 0" ::: "memory");  // C-layout -> A-layout flip

    const v16h pf = load_a_frag(Pw, LDS_STR);
#pragma unroll
    for (int dt = 0; dt < 4; ++dt) {
      const v16h vb = load_b_frag(vt + ((size_t)bh * 64 + dt * 16) * 2048 + c0, 2048);
      o[dt] = wmma16(pf, vb, o[dt]);
    }
  }

  float inv[8];
#pragma unroll
  for (int r = 0; r < 8; ++r) inv[r] = 1.0f / l[r];
#pragma unroll
  for (int dt = 0; dt < 4; ++dt)
#pragma unroll
    for (int r = 0; r < 8; ++r) {
      const int n = rBase + r + hal * 8;
      const int d = dt * 16 + ln;
      inner[((size_t)b * 2048 + n) * 1024 + h * 64 + d] = (f16)(o[dt][r] * inv[r]);
    }
}

// ---------------------------------------------------------------------------
// K3: out GEMM + bias -> fp32
// ---------------------------------------------------------------------------
__global__ __launch_bounds__(256)
void out_gemm_kernel(const f16* __restrict__ A, const f16* __restrict__ BT,
                     const float* __restrict__ bias, float* __restrict__ out) {
  __shared__ f16 smem[4 * 128 * LDS_STR];
  const int lane = threadIdx.x & 31;
  const int wave = threadIdx.x >> 5;
  const int wm = wave & 3, wn = wave >> 2;
  const int mBase = blockIdx.y * BM, nBase = blockIdx.x * BN;

  v8f acc[2][4] = {};
  gemm_loop(A, BT, smem, mBase, nBase, acc);

  const int hal = lane >> 4, ln = lane & 15;
#pragma unroll
  for (int sm = 0; sm < 2; ++sm)
#pragma unroll
    for (int sn = 0; sn < 4; ++sn)
#pragma unroll
      for (int r = 0; r < 8; ++r) {
        const int grow = mBase + wm * 32 + sm * 16 + r + hal * 8;
        const int gcol = nBase + wn * 64 + sn * 16 + ln;
        out[(size_t)grow * 1024 + gcol] = acc[sm][sn][r] + bias[gcol];
      }
}

// ---------------------------------------------------------------------------
extern "C" void kernel_launch(void* const* d_in, const int* in_sizes, int n_in,
                              void* d_out, int out_size, void* d_ws,
                              size_t ws_size, hipStream_t stream) {
  const float* x     = (const float*)d_in[0];  // [4,2048,1024]
  const float* w_qkv = (const float*)d_in[1];  // [1024,3072]
  const float* w_out = (const float*)d_in[2];  // [1024,1024]
  const float* b_out = (const float*)d_in[3];  // [1024]
  float* out = (float*)d_out;                  // [4,2048,1024]

  char* ws = (char*)d_ws;
  size_t off = 0;
  f16* xh    = (f16*)(ws + off); off += (size_t)8192 * 1024 * 2;  // 16 MB
  f16* wqT   = (f16*)(ws + off); off += (size_t)3072 * 1024 * 2;  //  6 MB
  f16* woT   = (f16*)(ws + off); off += (size_t)1024 * 1024 * 2;  //  2 MB
  f16* qb    = (f16*)(ws + off); off += (size_t)64 * 2048 * 64 * 2;
  f16* kb    = (f16*)(ws + off); off += (size_t)64 * 2048 * 64 * 2;
  f16* vtb   = (f16*)(ws + off); off += (size_t)64 * 2048 * 64 * 2;
  f16* inner = (f16*)(ws + off); off += (size_t)8192 * 1024 * 2;

  cvt_kernel<<<8192, 256, 0, stream>>>(x, xh);
  transpose_cvt_kernel<<<dim3(96, 32), 256, 0, stream>>>(w_qkv, wqT, 1024, 3072);
  transpose_cvt_kernel<<<dim3(32, 32), 256, 0, stream>>>(w_out, woT, 1024, 1024);

  qkv_gemm_kernel<<<dim3(24, 64), 256, 0, stream>>>(xh, wqT, qb, kb, vtb);
  attn_kernel    <<<dim3(64, 16), 256, 0, stream>>>(qb, kb, vtb, inner);
  out_gemm_kernel<<<dim3(8, 64),  256, 0, stream>>>(inner, woT, b_out, out);
}
